// RGCNEncoder_25984552141045
// MI455X (gfx1250) — matrix operand order repaired and checked by reading, not compiled, not run on hardware
//
#include <hip/hip_runtime.h>

// RGCN encoder, MI455X (gfx1250, wave32, WMMA).
// Only layer L-1 matters (reference reads self.embeddings each layer and
// overwrites output), and source==dest per edge slot, so:
//   out[n] = (sum_r c[n,r] * (emb[n] @ W2_r + b2_r)) / sum_r c[n,r]
// c[n,r] = edge-slot histogram (18 relation-direction slots).
// Compute: V_WMMA_F32_16X16X32_BF16, f32 accumulate.
// Weights (147 KB bf16) are staged into LDS once per block via
// GLOBAL_LOAD_ASYNC_TO_LDS_B128 (ASYNCcnt), hot loop feeds WMMA from LDS.

typedef __bf16 v16bf __attribute__((ext_vector_type(16)));
typedef float  v8f   __attribute__((ext_vector_type(8)));

#define HID    64
#define NRELD  18      // 2 * 9 relation-direction slots
#define RHALF  9
#define TILE_M 16
#define WPB    8       // waves per block (256 threads, wave32)
#define WFRAG  (NRELD * 8 * 512)   // 73728 ushorts = 147456 B

// ---- float -> bf16, round-to-nearest-even, no libcalls ----
__device__ __forceinline__ unsigned short f2bf(float f) {
    unsigned int u = __float_as_uint(f);
    u = (u + 0x7FFFu + ((u >> 16) & 1u)) >> 16;
    return (unsigned short)u;
}

// ---- 1) zero the (node, rel-slot) counter array ----
__global__ __launch_bounds__(256) void zero_u32(unsigned int* p, size_t n) {
    size_t i = (size_t)blockIdx.x * blockDim.x + threadIdx.x;
    size_t stride = (size_t)gridDim.x * blockDim.x;
    for (; i < n; i += stride) p[i] = 0u;
}

// ---- 2) edge-slot histogram: c[idx0,rel] += 1 ; c[idx1,rel+9] += 1 ----
__global__ __launch_bounds__(256) void count_edges(
    const int* __restrict__ ei, const int* __restrict__ et,
    unsigned int* __restrict__ cnt, int E) {
    int i = blockIdx.x * blockDim.x + threadIdx.x;
    int stride = gridDim.x * blockDim.x;
    for (; i < E; i += stride) {
        int i0 = ei[i];
        int i1 = ei[(size_t)E + i];
        int r  = et[i];
        atomicAdd(&cnt[(size_t)i0 * NRELD + r], 1u);
        atomicAdd(&cnt[(size_t)i1 * NRELD + RHALF + r], 1u);
    }
}

// ---- 3) pack weights[L-1] (18 x 64 x 64 f32, [r][k][n]) into bf16 B-frags.
// Fragment (r, kt, nt) is a 32x16 bf16 B tile; per ISA B layout lane holds
// column N=lane&15, K = 8h + e (e<8), 16 + 8h + (e-8) (e>=8), h = lane>>4.
__global__ __launch_bounds__(256) void pack_weights(
    const float* __restrict__ w2, unsigned short* __restrict__ wbf) {
    int idx = blockIdx.x * blockDim.x + threadIdx.x;     // over 18*64*64
    if (idx >= NRELD * HID * HID) return;
    int r  = idx / (HID * HID);
    int k  = (idx / HID) % HID;
    int n  = idx % HID;
    int kt = k >> 5, kk = k & 31;
    int nt = n >> 4, nn = n & 15;
    int lane = (((kk & 15) >> 3) << 4) + nn;     // ((kk%16)/8)*16 + nn
    int e    = (kk & 7) + ((kk >> 4) << 3);
    size_t dst = ((size_t)((r * 2 + kt) * 4 + nt)) * 512 + lane * 16 + e;
    wbf[dst] = f2bf(w2[idx]);
}

// ---- 4) pack embeddings into bf16 A-frags (ISA 16-bit A 16x32 layout).
// Within K-chunk kt, lane-half h holds K in {8h..8h+7} (e=0..7) and
// {16+8h..16+8h+7} (e=8..15). Store so each lane reads 16 contiguous bf16.
__global__ __launch_bounds__(256) void pack_embeddings(
    const float* __restrict__ emb, unsigned short* __restrict__ abf, size_t n) {
    size_t i = (size_t)blockIdx.x * blockDim.x + threadIdx.x;
    size_t stride = (size_t)gridDim.x * blockDim.x;
    for (; i < n; i += stride) {
        size_t node = i >> 6;
        int k  = (int)(i & 63);
        int kt = k >> 5, kk = k & 31;
        int h, e;
        if (kk < 16) { h = kk >> 3;        e = kk & 7; }
        else         { h = (kk - 16) >> 3; e = 8 + ((kk - 16) & 7); }
        abf[node * HID + kt * 32 + h * 16 + e] = f2bf(emb[i]);
    }
}

// ---- 5) main: per wave one 16-node tile; 18 rels x 4 ntiles x 2 K-steps ----
__global__ __launch_bounds__(256) void rgcn_wmma_kernel(
    const unsigned short* __restrict__ abf,
    const unsigned short* __restrict__ wbf,
    const unsigned int*   __restrict__ cnt,
    const float*          __restrict__ bias2,
    float*                __restrict__ out,
    int nNodes) {
    __shared__ unsigned short sW[WFRAG];                   // 147.5 KB
    __shared__ float          sBias[NRELD * HID];          // 4.6 KB
    __shared__ unsigned int   sCnt[WPB * TILE_M * NRELD];  // 9.2 KB

    const int tid    = threadIdx.x;
    const int nTiles = (nNodes + TILE_M - 1) / TILE_M;
    const int tile0  = blockIdx.x * WPB;

    // --- async-stage the whole bf16 weight fragment block into LDS ---
    // 147456 B / (256 threads * 16 B) = 36 async b128 copies per thread.
    {
        unsigned ldsBase = (unsigned)(size_t)(void*)&sW[0];
        const char* gbase = (const char*)wbf;
#pragma unroll 4
        for (int it = 0; it < 36; ++it) {
            unsigned byteOff = (unsigned)(it * 256 + tid) * 16u;
            unsigned lds  = ldsBase + byteOff;
            asm volatile("global_load_async_to_lds_b128 %0, %1, %2"
                         :
                         : "v"(lds), "v"(byteOff), "s"(gbase)
                         : "memory");
        }
    }

    // --- overlap: stage bias + per-tile counts with normal loads ---
    for (int i = tid; i < NRELD * HID; i += 256) sBias[i] = bias2[i];
    for (int i = tid; i < WPB * TILE_M * NRELD; i += 256) {
        int w    = i / (TILE_M * NRELD);
        int rem  = i - w * (TILE_M * NRELD);
        int row  = rem / NRELD;
        int r    = rem - row * NRELD;
        int node = (tile0 + w) * TILE_M + row;
        sCnt[i] = (node < nNodes) ? cnt[(size_t)node * NRELD + r] : 0u;
    }

    asm volatile("s_wait_asynccnt 0" ::: "memory");
    __syncthreads();

    const int wave = tid >> 5, lane = tid & 31;
    const int tile = tile0 + wave;
    if (tile >= nTiles) return;
    const int base = tile * TILE_M;
    const int h    = lane >> 4;    // lane half
    const int col  = lane & 15;    // C column N; also A row M for loads

    // A fragments: 16 contiguous bf16 per lane, 32B aligned, from global.
    const v16bf A0 = *reinterpret_cast<const v16bf*>(
        abf + (size_t)(base + col) * HID + h * 16);
    const v16bf A1 = *reinterpret_cast<const v16bf*>(
        abf + (size_t)(base + col) * HID + 32 + h * 16);

    v8f acc[4] = {v8f{}, v8f{}, v8f{}, v8f{}};
    v8f totv{};
    const unsigned int* cw = &sCnt[wave * TILE_M * NRELD];

#pragma unroll 2
    for (int r = 0; r < NRELD; ++r) {
        // per-row counts: C row M = v + 8*h
        v8f cf;
#pragma unroll
        for (int v = 0; v < 8; ++v)
            cf[v] = (float)cw[(v + 8 * h) * NRELD + r];
        totv += cf;

        const unsigned short* wr = &sW[r * 8 * 512];
#pragma unroll
        for (int nt = 0; nt < 4; ++nt) {
            v16bf B0 = *reinterpret_cast<const v16bf*>(wr + (0 * 4 + nt) * 512 + lane * 16);
            v16bf B1 = *reinterpret_cast<const v16bf*>(wr + (1 * 4 + nt) * 512 + lane * 16);
            v8f y{};
            y = __builtin_amdgcn_wmma_f32_16x16x32_bf16(false, A0, false, B0,
                                                        (short)0, y, false, false);
            y = __builtin_amdgcn_wmma_f32_16x16x32_bf16(false, A1, false, B1,
                                                        (short)0, y, false, false);
            float bv = sBias[r * HID + nt * 16 + col];
            acc[nt] += cf * (y + bv);
        }
    }

    // scatter_mean epilogue: divide by total incident count, 0 if none.
#pragma unroll
    for (int nt = 0; nt < 4; ++nt) {
#pragma unroll
        for (int v = 0; v < 8; ++v) {
            int M = v + 8 * h;
            int node = base + M;
            if (node < nNodes) {
                float t = totv[v];
                float o = (t > 0.0f) ? acc[nt][v] / t : 0.0f;
                out[(size_t)node * HID + nt * 16 + col] = o;
            }
        }
    }
}

extern "C" void kernel_launch(void* const* d_in, const int* in_sizes, int n_in,
                              void* d_out, int out_size, void* d_ws, size_t ws_size,
                              hipStream_t stream) {
    const int*   ei  = (const int*)d_in[0];      // edge_index [2, E]
    const int*   et  = (const int*)d_in[1];      // edge_type  [E]
    const float* emb = (const float*)d_in[2];    // [N, 64]
    const float* wts = (const float*)d_in[3];    // [L, 18, 64, 64]
    const float* bia = (const float*)d_in[4];    // [L, 18, 64]

    const int E = in_sizes[1];
    const int N = in_sizes[2] / HID;
    const int L = in_sizes[3] / (NRELD * HID * HID);
    const float* w2 = wts + (size_t)(L - 1) * NRELD * HID * HID;
    const float* b2 = bia + (size_t)(L - 1) * NRELD * HID;

    // workspace layout (256B aligned sections)
    size_t cntBytes = (size_t)N * NRELD * sizeof(unsigned int);
    size_t wOff     = (cntBytes + 255) & ~(size_t)255;
    size_t wBytes   = (size_t)WFRAG * sizeof(unsigned short);
    size_t aOff     = (wOff + wBytes + 255) & ~(size_t)255;
    size_t aBytes   = (size_t)N * HID * sizeof(unsigned short);
    if (ws_size < aOff + aBytes) return;   // ~20.2 MB needed

    unsigned int*   cnt = (unsigned int*)((char*)d_ws);
    unsigned short* wbf = (unsigned short*)((char*)d_ws + wOff);
    unsigned short* abf = (unsigned short*)((char*)d_ws + aOff);
    float*          out = (float*)d_out;

    zero_u32<<<512, 256, 0, stream>>>(cnt, (size_t)N * NRELD);
    count_edges<<<1024, 256, 0, stream>>>(ei, et, cnt, E);
    pack_weights<<<(NRELD * HID * HID + 255) / 256, 256, 0, stream>>>(w2, wbf);
    pack_embeddings<<<2048, 256, 0, stream>>>(emb, abf, (size_t)N * HID);

    int nTiles  = (N + TILE_M - 1) / TILE_M;
    int nBlocks = (nTiles + WPB - 1) / WPB;
    rgcn_wmma_kernel<<<nBlocks, 256, 0, stream>>>(abf, wbf, cnt, b2, out, N);
}
